// TrajectoryS4Model_18253611008311
// MI455X (gfx1250) — matrix operand order repaired and checked by reading, not compiled, not run on hardware
//
#include <hip/hip_runtime.h>
#include <hip/hip_bf16.h>

// ---------------- model constants ----------------
#define B_    64
#define L_    250
#define SD_   17
#define AD_   6
#define H_    384
#define H2_   768
#define NL_   12
#define NS_   32          // complex modes per channel
#define LK_   1000        // 4*L

// All activations stored channel-contiguous: (B, Lk, H)

typedef __attribute__((ext_vector_type(16))) __bf16 v16bf;
typedef __attribute__((ext_vector_type(8)))  __bf16 bf16x8;
typedef __attribute__((ext_vector_type(8)))  float  v8f;

// =================================================================
// K0: embedding + interleave  ->  X (B, Lk, H) fp32
// consecutive threads = consecutive h -> all 4 stores coalesced
// =================================================================
__global__ void embed_kernel(const float* __restrict__ st, const float* __restrict__ ac,
                             const float* __restrict__ rtg, const float* __restrict__ ctg,
                             const int*   __restrict__ ts,
                             const float* __restrict__ Ws,  const float* __restrict__ bs,
                             const float* __restrict__ Wa,  const float* __restrict__ ba,
                             const float* __restrict__ Wr,  const float* __restrict__ brn,
                             const float* __restrict__ Wc,  const float* __restrict__ bc,
                             const float* __restrict__ Et,  float* __restrict__ X)
{
    int idx = blockIdx.x * blockDim.x + threadIdx.x;   // ((b*L + l)*H + h)
    if (idx >= B_ * L_ * H_) return;
    int hh = idx % H_;
    int bl = idx / H_;
    int b  = bl / L_;
    int l  = bl % L_;

    float te = Et[(size_t)ts[bl] * H_ + hh];
    float re = fmaf(rtg[bl], Wr[hh], brn[hh] + te);
    float ce = fmaf(ctg[bl], Wc[hh], bc[hh] + te);

    float se = bs[hh] + te;
    const float* sp = st + (size_t)bl * SD_;
    const float* wr = Ws + (size_t)hh * SD_;
#pragma unroll
    for (int i = 0; i < SD_; ++i) se = fmaf(sp[i], wr[i], se);

    float ae = ba[hh] + te;
    const float* ap = ac + (size_t)bl * AD_;
    const float* wa = Wa + (size_t)hh * AD_;
#pragma unroll
    for (int i = 0; i < AD_; ++i) ae = fmaf(ap[i], wa[i], ae);

    size_t base = ((size_t)b * LK_ + 4 * l) * H_ + hh;
    X[base]          = re;   // token order (return, constraint, state, action)
    X[base + H_]     = ce;
    X[base + 2*H_]   = se;
    X[base + 3*H_]   = ae;
}

// =================================================================
// P0: per-(layer,h,n) discretized SSM params: (lam_re, lam_im, 2C'_re, 2C'_im)
// =================================================================
__global__ void prep_params(const float* __restrict__ log_dt, const float* __restrict__ log_Ar,
                            const float* __restrict__ Aim,   const float* __restrict__ Cre,
                            const float* __restrict__ Cim,   float4* __restrict__ P)
{
    int idx = blockIdx.x * blockDim.x + threadIdx.x;
    if (idx >= NL_ * H_ * NS_) return;
    int h     = (idx / NS_) % H_;
    int layer = idx / (NS_ * H_);

    float dt  = __expf(log_dt[layer * H_ + h]);
    float Are = -__expf(log_Ar[idx]);
    float Ai  = Aim[idx];
    float xr  = Are * dt, xi = Ai * dt;
    float er  = __expf(xr);
    float lr  = er * __cosf(xi);
    float li  = er * __sinf(xi);
    // C' = (Cre + i Cim) * (lam - 1) / A   (complex division via conj)
    float nr  = lr - 1.f, ni = li;
    float inv = 1.f / (Are * Are + Ai * Ai);
    float qr  = (nr * Are + ni * Ai) * inv;
    float qi  = (ni * Are - nr * Ai) * inv;
    float cr  = Cre[idx] * qr - Cim[idx] * qi;
    float ci  = Cre[idx] * qi + Cim[idx] * qr;
    P[idx] = make_float4(lr, li, 2.f * cr, 2.f * ci);
}

// =================================================================
// P1: W_out fp32 -> bf16 (all layers)
// =================================================================
__global__ void prep_w(const float* __restrict__ Wo, __bf16* __restrict__ Wb, int n)
{
    int idx = blockIdx.x * blockDim.x + threadIdx.x;
    if (idx < n) Wb[idx] = (__bf16)Wo[idx];
}

// =================================================================
// K1: S4D diagonal recurrence + D-skip + exact GELU -> Y bf16 (B,Lk,H)
// Mode-parallel split: each (b,h) channel is handled by SCAN_MG=4 threads,
// 8 complex modes each (light registers -> high occupancy, 4x wave count).
// Partials combined every SCAN_T=8 steps via double-buffered LDS
// (one barrier per chunk). All global traffic coalesced across channels.
// =================================================================
#define SCAN_CH  96      // channels per block
#define SCAN_MG  4       // mode groups per channel
#define SCAN_T   8       // timesteps per chunk  (125 chunks)
#define SCAN_PAD 12      // padded row (floats) -> float4-aligned, conflict-light

__global__ __launch_bounds__(SCAN_CH * SCAN_MG)
void scan_kernel(const float* __restrict__ X,
                 const float4* __restrict__ P,   // layer base (H*NS)
                 const float*  __restrict__ Dp,  // layer base (H)
                 __bf16* __restrict__ Y)
{
    __shared__ float pbuf[2][SCAN_MG][SCAN_CH][SCAN_PAD];

    const int b   = blockIdx.x;
    const int h0  = blockIdx.y * SCAN_CH;
    const int tid = threadIdx.x;
    const int hc  = tid % SCAN_CH;     // lanes = consecutive channels
    const int mg  = tid / SCAN_CH;     // mode group 0..3
    const int hh  = h0 + hc;

    float lr[8], li[8], cr[8], ci[8], sr[8], si[8];
    const float4* pp = P + (size_t)hh * NS_ + mg * 8;
#pragma unroll
    for (int n = 0; n < 8; ++n) {
        float4 q = pp[n];
        lr[n] = q.x; li[n] = q.y; cr[n] = q.z; ci[n] = q.w;
        sr[n] = 0.f; si[n] = 0.f;
    }
    const float Dh = Dp[hh];
    const float* xp = X + (size_t)b * LK_ * H_ + hh;
    __bf16*      yp = Y + (size_t)b * LK_ * H_ + hh;

    int pb = 0;
    for (int t0 = 0; t0 < LK_; t0 += SCAN_T) {
        float xa[SCAN_T], pt[SCAN_T];
#pragma unroll
        for (int j = 0; j < SCAN_T; ++j) xa[j] = xp[(size_t)(t0 + j) * H_];
#pragma unroll
        for (int j = 0; j < SCAN_T; ++j) {
            float acc = 0.f;
#pragma unroll
            for (int n = 0; n < 8; ++n) {
                float nre = fmaf(lr[n], sr[n], fmaf(-li[n], si[n], xa[j]));  // lam*s + x
                float nim = fmaf(li[n], sr[n], lr[n] * si[n]);
                sr[n] = nre; si[n] = nim;
                acc = fmaf(cr[n], nre, fmaf(-ci[n], nim, acc));              // 2*Re(C'·s)
            }
            pt[j] = acc;
        }
        // publish partials (two aligned float4 LDS stores)
        float4* dst = (float4*)&pbuf[pb][mg][hc][0];
        dst[0] = make_float4(pt[0], pt[1], pt[2], pt[3]);
        dst[1] = make_float4(pt[4], pt[5], pt[6], pt[7]);
        __syncthreads();
        // each mode group finalizes 2 of the 8 timesteps
#pragma unroll
        for (int u = 0; u < 2; ++u) {
            int j = mg * 2 + u;
            float ysum = pbuf[pb][0][hc][j] + pbuf[pb][1][hc][j]
                       + pbuf[pb][2][hc][j] + pbuf[pb][3][hc][j];
            float y = fmaf(Dh, xa[j], ysum);
            y = 0.5f * y * (1.f + erff(y * 0.70710678118f));                 // exact GELU
            yp[(size_t)(t0 + j) * H_] = (__bf16)y;
        }
        pb ^= 1;   // double buffer: next chunk's writes can't race these reads
    }
}

// =================================================================
// K2: GLU GEMM via WMMA bf16, LDS-free / barrier-free.
// Each wave: 16-row (lo+hi GLU pair) x 32-col tile -> A fragments reused
// across two B fragments: 8 b128 loads per 4 WMMAs. K fully unrolled.
// fused: +bias, GLU, +residual -> XR fp32 (B,Lk,H)
// =================================================================
__global__ __launch_bounds__(256) void glu_kernel(const __bf16* __restrict__ Wb,  // layer (768,384)
                                                  const __bf16* __restrict__ Yb,  // (B,Lk,H)
                                                  const float*  __restrict__ Xin, // residual (B,Lk,H)
                                                  const float*  __restrict__ bo,  // layer (768)
                                                  float* __restrict__ XR)
{
    const int tid  = threadIdx.x;
    const int w    = tid >> 5;             // wave id (wave32)
    const int lane = tid & 31;
    const int b    = blockIdx.z;
    const int m0   = blockIdx.y * 16;
    const int n0   = blockIdx.x * 256;     // 8 waves x 32 cols

    const int mrow = lane & 15;
    const int khi  = lane >> 4;

    const int ncol0 = n0 + w * 32 + (lane & 15);
    const int ncol1 = ncol0 + 16;
    const int g0 = (ncol0 < LK_) ? ncol0 : (LK_ - 1);   // clamp keeps loads in-bounds
    const int g1 = (ncol1 < LK_) ? ncol1 : (LK_ - 1);

    // A fragment rows (lo / hi GLU halves), K contiguous in memory
    const __bf16* arow_lo = Wb + (size_t)(m0 + mrow) * H_;
    const __bf16* arow_hi = Wb + (size_t)(H_ + m0 + mrow) * H_;
    // B fragment columns: fixed seq position, 16 consecutive channels
    const __bf16* brow0 = Yb + ((size_t)b * LK_ + g0) * H_;
    const __bf16* brow1 = Yb + ((size_t)b * LK_ + g1) * H_;

    v8f accL0 = {}, accH0 = {}, accL1 = {}, accH1 = {};

#pragma unroll
    for (int k0 = 0; k0 < H_; k0 += 32) {
        // A 16x32 layout: lane(0-15)=M; halves = K {16khi .. 16khi+15} contiguous
        bf16x8 a0 = *(const bf16x8*)(arow_lo + k0 + khi * 16);
        bf16x8 a1 = *(const bf16x8*)(arow_lo + k0 + khi * 16 + 8);
        bf16x8 a2 = *(const bf16x8*)(arow_hi + k0 + khi * 16);
        bf16x8 a3 = *(const bf16x8*)(arow_hi + k0 + khi * 16 + 8);
        // B 32x16 layout: N = lane&15; halves = K {16khi .. 16khi+15} contiguous
        bf16x8 b0 = *(const bf16x8*)(brow0 + k0 + khi * 16);
        bf16x8 b1 = *(const bf16x8*)(brow0 + k0 + khi * 16 + 8);
        bf16x8 b2 = *(const bf16x8*)(brow1 + k0 + khi * 16);
        bf16x8 b3 = *(const bf16x8*)(brow1 + k0 + khi * 16 + 8);

        v16bf af0, af1, bf0, bf1;
#pragma unroll
        for (int i = 0; i < 8; ++i) {
            af0[i] = a0[i]; af0[8 + i] = a1[i];
            af1[i] = a2[i]; af1[8 + i] = a3[i];
            bf0[i] = b0[i]; bf0[8 + i] = b1[i];
            bf1[i] = b2[i]; bf1[8 + i] = b3[i];
        }

        accL0 = __builtin_amdgcn_wmma_f32_16x16x32_bf16(false, af0, false, bf0,
                                                        (short)0, accL0, false, false);
        accH0 = __builtin_amdgcn_wmma_f32_16x16x32_bf16(false, af1, false, bf0,
                                                        (short)0, accH0, false, false);
        accL1 = __builtin_amdgcn_wmma_f32_16x16x32_bf16(false, af0, false, bf1,
                                                        (short)0, accL1, false, false);
        accH1 = __builtin_amdgcn_wmma_f32_16x16x32_bf16(false, af1, false, bf1,
                                                        (short)0, accH1, false, false);
    }

    // ---- epilogue: bias + GLU + residual; 8 contiguous channels per lane ----
    if (ncol0 < LK_) {
        size_t cb = ((size_t)b * LK_ + ncol0) * H_;
#pragma unroll
        for (int r = 0; r < 8; ++r) {
            int   mg = m0 + r + 8 * khi;      // C/D layout: VGPR r -> M = r + 8*khi
            float zl = accL0[r] + bo[mg];
            float zh = accH0[r] + bo[H_ + mg];
            float g  = zl * (1.f / (1.f + __expf(-zh)));
            XR[cb + mg] = g + Xin[cb + mg];
        }
    }
    if (ncol1 < LK_) {
        size_t cb = ((size_t)b * LK_ + ncol1) * H_;
#pragma unroll
        for (int r = 0; r < 8; ++r) {
            int   mg = m0 + r + 8 * khi;
            float zl = accL1[r] + bo[mg];
            float zh = accH1[r] + bo[H_ + mg];
            float g  = zl * (1.f / (1.f + __expf(-zh)));
            XR[cb + mg] = g + Xin[cb + mg];
        }
    }
}

// =================================================================
// K3: LayerNorm over channel dim H per (b,l) column.
// (B,Lk,H) layout -> column is 384 contiguous floats. One wave per column,
// 12 values/lane (3x float4), 5-step shfl_xor butterfly reduction.
// =================================================================
__global__ __launch_bounds__(256) void ln_kernel(const float* __restrict__ XR,
                                                 const float* __restrict__ lw,
                                                 const float* __restrict__ lb,
                                                 float* __restrict__ X)
{
    const int w    = threadIdx.x >> 5;
    const int lane = threadIdx.x & 31;
    const size_t col = (size_t)blockIdx.x * 8 + w;     // b*Lk + l, < 64000

    const float* cp = XR + col * H_ + lane * 12;
    float4 v0 = *(const float4*)(cp + 0);
    float4 v1 = *(const float4*)(cp + 4);
    float4 v2 = *(const float4*)(cp + 8);

    float s  = v0.x + v0.y + v0.z + v0.w + v1.x + v1.y + v1.z + v1.w
             + v2.x + v2.y + v2.z + v2.w;
    float s2 = v0.x*v0.x + v0.y*v0.y + v0.z*v0.z + v0.w*v0.w
             + v1.x*v1.x + v1.y*v1.y + v1.z*v1.z + v1.w*v1.w
             + v2.x*v2.x + v2.y*v2.y + v2.z*v2.z + v2.w*v2.w;
#pragma unroll
    for (int m = 16; m >= 1; m >>= 1) {
        s  += __shfl_xor(s,  m, 32);
        s2 += __shfl_xor(s2, m, 32);
    }
    float mu  = s * (1.f / H_);
    float var = s2 * (1.f / H_) - mu * mu;
    float rs  = rsqrtf(var + 1e-5f);

    const float4* wv = (const float4*)(lw + lane * 12);
    const float4* bv = (const float4*)(lb + lane * 12);
    float4 w0 = wv[0], w1 = wv[1], w2 = wv[2];
    float4 g0 = bv[0], g1 = bv[1], g2 = bv[2];

    float* op = X + col * H_ + lane * 12;
    float4 o0, o1, o2;
    o0.x = (v0.x - mu) * rs * w0.x + g0.x;  o0.y = (v0.y - mu) * rs * w0.y + g0.y;
    o0.z = (v0.z - mu) * rs * w0.z + g0.z;  o0.w = (v0.w - mu) * rs * w0.w + g0.w;
    o1.x = (v1.x - mu) * rs * w1.x + g1.x;  o1.y = (v1.y - mu) * rs * w1.y + g1.y;
    o1.z = (v1.z - mu) * rs * w1.z + g1.z;  o1.w = (v1.w - mu) * rs * w1.w + g1.w;
    o2.x = (v2.x - mu) * rs * w2.x + g2.x;  o2.y = (v2.y - mu) * rs * w2.y + g2.y;
    o2.z = (v2.z - mu) * rs * w2.z + g2.z;  o2.w = (v2.w - mu) * rs * w2.w + g2.w;
    *(float4*)(op + 0) = o0;
    *(float4*)(op + 4) = o1;
    *(float4*)(op + 8) = o2;
}

// =================================================================
// K4: final projections from state/action token streams.
// (B,Lk,H) layout -> each token row is 384 contiguous floats (float4 reads).
// =================================================================
__global__ void proj_kernel(const float* __restrict__ X,
                            const float* __restrict__ Wps, const float* __restrict__ bps,
                            const float* __restrict__ Wpa, const float* __restrict__ bpa,
                            float* __restrict__ out)
{
    int idx = blockIdx.x * blockDim.x + threadIdx.x;   // b*L + l
    if (idx >= B_ * L_) return;
    int b = idx / L_;
    int l = idx % L_;

    float accS[SD_], accA[AD_];
#pragma unroll
    for (int s = 0; s < SD_; ++s) accS[s] = bps[s];
#pragma unroll
    for (int s = 0; s < AD_; ++s) accA[s] = bpa[s];

    const float4* row2 = (const float4*)(X + ((size_t)b * LK_ + 4 * l + 2) * H_); // state tokens
    const float4* row3 = (const float4*)(X + ((size_t)b * LK_ + 4 * l + 3) * H_); // action tokens

    for (int i = 0; i < H_ / 4; ++i) {
        float4 x3 = row3[i];
        float4 x2 = row2[i];
#pragma unroll
        for (int s = 0; s < SD_; ++s) {
            float4 wv = *(const float4*)(Wps + (size_t)s * H_ + 4 * i);
            accS[s] = fmaf(x3.x, wv.x, fmaf(x3.y, wv.y, fmaf(x3.z, wv.z, fmaf(x3.w, wv.w, accS[s]))));
        }
#pragma unroll
        for (int s = 0; s < AD_; ++s) {
            float4 wv = *(const float4*)(Wpa + (size_t)s * H_ + 4 * i);
            accA[s] = fmaf(x2.x, wv.x, fmaf(x2.y, wv.y, fmaf(x2.z, wv.z, fmaf(x2.w, wv.w, accA[s]))));
        }
    }
    float* os = out + (size_t)idx * SD_;
#pragma unroll
    for (int s = 0; s < SD_; ++s) os[s] = accS[s];
    float* oa = out + (size_t)B_ * L_ * SD_ + (size_t)idx * AD_;
#pragma unroll
    for (int s = 0; s < AD_; ++s) oa[s] = accA[s];
}

// =================================================================
// host launcher
// =================================================================
extern "C" void kernel_launch(void* const* d_in, const int* in_sizes, int n_in,
                              void* d_out, int out_size, void* d_ws, size_t ws_size,
                              hipStream_t stream)
{
    // ---- inputs (setup_inputs dict order; layer_params flattened in insertion order) ----
    const float* states  = (const float*)d_in[0];
    const float* actions = (const float*)d_in[1];
    const float* rtg     = (const float*)d_in[2];
    const float* ctg     = (const float*)d_in[3];
    const int*   tsteps  = (const int*)  d_in[4];
    const float* W_state = (const float*)d_in[5];
    const float* b_state = (const float*)d_in[6];
    const float* W_act   = (const float*)d_in[7];
    const float* b_act   = (const float*)d_in[8];
    const float* W_ret   = (const float*)d_in[9];
    const float* b_ret   = (const float*)d_in[10];
    const float* W_con   = (const float*)d_in[11];
    const float* b_con   = (const float*)d_in[12];
    const float* E_time  = (const float*)d_in[13];
    const float* log_dt  = (const float*)d_in[14];   // (NL,H)
    const float* log_Ar  = (const float*)d_in[15];   // (NL,H,N)
    const float* A_imag  = (const float*)d_in[16];
    const float* C_re    = (const float*)d_in[17];
    const float* C_im    = (const float*)d_in[18];
    const float* Dparam  = (const float*)d_in[19];   // (NL,H)
    const float* W_out   = (const float*)d_in[20];   // (NL,2H,H)
    const float* b_out   = (const float*)d_in[21];   // (NL,2H)
    const float* ln_w    = (const float*)d_in[22];   // (NL,H)
    const float* ln_b    = (const float*)d_in[23];
    const float* W_ps    = (const float*)d_in[24];
    const float* b_ps    = (const float*)d_in[25];
    const float* W_pa    = (const float*)d_in[26];
    const float* b_pa    = (const float*)d_in[27];
    float* out = (float*)d_out;

    // ---- workspace carve-up ----
    char* w = (char*)d_ws;
    size_t off = 0;
    auto carve = [&](size_t bytes) -> char* {
        char* p = w + off;
        off += (bytes + 255) & ~(size_t)255;
        return p;
    };
    float*   X   = (float*)  carve((size_t)B_ * LK_ * H_ * sizeof(float));     // 98.3 MB
    float*   XR  = (float*)  carve((size_t)B_ * LK_ * H_ * sizeof(float));     // 98.3 MB
    __bf16*  Ybf = (__bf16*) carve((size_t)B_ * LK_ * H_ * sizeof(__bf16));    // 49.2 MB
    float4*  P4  = (float4*) carve((size_t)NL_ * H_ * NS_ * sizeof(float4));   // 2.4 MB
    __bf16*  Wbf = (__bf16*) carve((size_t)NL_ * H2_ * H_ * sizeof(__bf16));   // 7.1 MB
    (void)ws_size; (void)n_in; (void)in_sizes; (void)out_size;

    // ---- one-time prep ----
    {
        int n = B_ * L_ * H_;
        embed_kernel<<<(n + 255) / 256, 256, 0, stream>>>(
            states, actions, rtg, ctg, tsteps,
            W_state, b_state, W_act, b_act, W_ret, b_ret, W_con, b_con,
            E_time, X);
    }
    {
        int n = NL_ * H_ * NS_;
        prep_params<<<(n + 255) / 256, 256, 0, stream>>>(log_dt, log_Ar, A_imag, C_re, C_im, P4);
    }
    {
        int n = NL_ * H2_ * H_;
        prep_w<<<(n + 255) / 256, 256, 0, stream>>>(W_out, Wbf, n);
    }

    // ---- 12 S4 layers ----
    for (int layer = 0; layer < NL_; ++layer) {
        const float4* Pl  = P4 + (size_t)layer * H_ * NS_;
        const float*  Dl  = Dparam + (size_t)layer * H_;
        const __bf16* Wl  = Wbf + (size_t)layer * H2_ * H_;
        const float*  bol = b_out + (size_t)layer * H2_;
        const float*  lwl = ln_w + (size_t)layer * H_;
        const float*  lbl = ln_b + (size_t)layer * H_;

        scan_kernel<<<dim3(B_, H_ / SCAN_CH), SCAN_CH * SCAN_MG, 0, stream>>>(X, Pl, Dl, Ybf);
        glu_kernel<<<dim3((LK_ + 255) / 256, H_ / 16, B_), 256, 0, stream>>>(Wl, Ybf, X, bol, XR);
        ln_kernel<<<dim3((B_ * LK_) / 8), 256, 0, stream>>>(XR, lwl, lbl, X);
    }

    // ---- final projections ----
    {
        int n = B_ * L_;
        proj_kernel<<<(n + 255) / 256, 256, 0, stream>>>(X, W_ps, b_ps, W_pa, b_pa, out);
    }
}
